// DistanceRelativeBias_541165879817
// MI455X (gfx1250) — compile-verified, matrix operand order checked by reading, not run on hardware
//
#include <hip/hip_runtime.h>

#define BATCH 4
#define NPTS  1024
#define HID   64
#define HEADS 16

typedef __attribute__((ext_vector_type(16))) _Float16 v16h;
typedef __attribute__((ext_vector_type(8)))  float    v8f;
typedef __attribute__((ext_vector_type(4)))  float    v4f;

// K index held by half-slot q (0..15) of a lane, per CDNA5 16-bit A-matrix
// layout (CDNA5 ISA §7.12.2):
//   lanes  0-15: K = q<8 ? q   : q+8   -> {0..7, 16..23}
//   lanes 16-31: K = q<8 ? q+8 : q+16  -> {8..15, 24..31}
__device__ __forceinline__ int kmap(int q, int hiHalf) {
  int k = (q < 8) ? q : (q + 8);
  return k + (hiHalf ? 8 : 0);
}

__device__ __forceinline__ float fast_silu(float x) {
  // silu(x) = x * sigmoid(x) = x * rcp(1 + exp2(-x * log2(e)))
  // v_exp_f32 + v_rcp_f32: avoids the IEEE div_scale/div_fmas/div_fixup chain.
  float e = __builtin_amdgcn_exp2f(x * -1.44269504088896340736f);
  return x * __builtin_amdgcn_rcpf(1.0f + e);
}

__global__ __launch_bounds__(256)
void distance_bias_wmma(const float* __restrict__ coords,
                        const float* __restrict__ w1,
                        const float* __restrict__ b1,
                        const float* __restrict__ w2,
                        const float* __restrict__ b2,
                        float* __restrict__ out)
{
  const int lane = threadIdx.x & 31;
  const int wave = blockIdx.x * (blockDim.x >> 5) + (threadIdx.x >> 5);

  const int chunk = wave & 3;                 // 4 chunks of 256 j-columns
  const int i     = (wave >> 2) & (NPTS - 1); // wave-uniform row
  const int b     = wave >> 12;               // batch
  const int hi    = lane >> 4;                // lane half
  const int lm    = lane & 15;                // A/D row id; B/D head id

  // ---- per-lane K-slices of w1/b1 (fixed pattern, hoisted) ----
  float w1r[32], b1r[32];
  #pragma unroll
  for (int q = 0; q < 16; ++q) {
    const int k = kmap(q, hi);
    w1r[q]      = w1[k];        b1r[q]      = b1[k];
    w1r[q + 16] = w1[k + 32];   b1r[q + 16] = b1[k + 32];
  }

  // ---- B matrix: w2 [64x16] as two f16 16x32-layout operands, N = lm ----
  v16h blo, bhi;
  #pragma unroll
  for (int q = 0; q < 16; ++q) {
    const int k = kmap(q, hi);
    blo[q] = (_Float16)w2[k * HEADS + lm];
    bhi[q] = (_Float16)w2[(k + 32) * HEADS + lm];
  }

  const float bias2 = b2[lm];

  // row-i coordinates (wave-uniform -> scalar loads)
  const float cix = coords[(b * NPTS + i) * 3 + 0];
  const float ciy = coords[(b * NPTS + i) * 3 + 1];
  const float ciz = coords[(b * NPTS + i) * 3 + 2];

  float* outBase = out + ((size_t)(b * HEADS + lm) * NPTS + i) * NPTS;

  for (int t = 0; t < 16; ++t) {
    const int j0 = chunk * 256 + t * 16;
    const int j  = j0 + lm;  // this lane's A-row M = lm maps to column j

    const float dx = cix - coords[(b * NPTS + j) * 3 + 0];
    const float dy = ciy - coords[(b * NPTS + j) * 3 + 1];
    const float dz = ciz - coords[(b * NPTS + j) * 3 + 2];
    const float d  = __builtin_sqrtf(dx * dx + dy * dy + dz * dz);

    // A matrix: h = silu(d * w1 + b1), K split 0..31 / 32..63
    v16h alo, ahi;
    #pragma unroll
    for (int q = 0; q < 16; ++q) {
      const float x0 = __builtin_fmaf(d, w1r[q],      b1r[q]);
      const float x1 = __builtin_fmaf(d, w1r[q + 16], b1r[q + 16]);
      alo[q] = (_Float16)fast_silu(x0);
      ahi[q] = (_Float16)fast_silu(x1);
    }

    v8f c;
    #pragma unroll
    for (int r = 0; r < 8; ++r) c[r] = bias2;   // + b2[head] init

    c = __builtin_amdgcn_wmma_f32_16x16x32_f16(false, alo, false, blo,
                                               (short)0, c, false, false);
    c = __builtin_amdgcn_wmma_f32_16x16x32_f16(false, ahi, false, bhi,
                                               (short)0, c, false, false);

    // D layout: VGPR r -> (M = r + 8*hi, N = lm). Lane's 8 floats are
    // contiguous in j -> two 16B stores per lane per tile.
    float* p = outBase + j0 + 8 * hi;
    v4f s0 = { c[0], c[1], c[2], c[3] };
    v4f s1 = { c[4], c[5], c[6], c[7] };
    *(v4f*)(p)     = s0;
    *(v4f*)(p + 4) = s1;
  }
}

extern "C" void kernel_launch(void* const* d_in, const int* in_sizes, int n_in,
                              void* d_out, int out_size, void* d_ws, size_t ws_size,
                              hipStream_t stream) {
  const float* coords = (const float*)d_in[0];
  const float* w1     = (const float*)d_in[1];
  const float* b1     = (const float*)d_in[2];
  const float* w2     = (const float*)d_in[3];
  const float* b2     = (const float*)d_in[4];
  float* out          = (float*)d_out;

  // total waves = BATCH * NPTS * 4 chunks = 16384; 8 waves per 256-thread block
  const int blocks = (BATCH * NPTS * 4) / 8;
  distance_bias_wmma<<<blocks, 256, 0, stream>>>(coords, w1, b1, w2, b2, out);
}
